// TransformerBiasEncoderLayer_35175782154705
// MI455X (gfx1250) — matrix operand order repaired and checked by reading, not compiled, not run on hardware
//
#include <hip/hip_runtime.h>
#include <hip/hip_bf16.h>

typedef unsigned short u16;
typedef __attribute__((ext_vector_type(16))) __bf16 bf16x16;
typedef __attribute__((ext_vector_type(8)))  float  f32x8;
typedef __attribute__((ext_vector_type(8)))  u16    u16x8;
typedef __attribute__((ext_vector_type(16))) u16    u16x16;
typedef __attribute__((ext_vector_type(4)))  unsigned int u32x4;
typedef __attribute__((ext_vector_type(8)))  int    i32x8;
typedef __attribute__((ext_vector_type(4)))  int    i32x4;

#define B_  32
#define L_  256
#define D_  1024
#define H_  8
#define DQ_ 128
#define DS_ 64
#define DM_ 256
#define DF_ 4096

__device__ __forceinline__ u16 f2bf(float f) {
    unsigned int u = __builtin_bit_cast(unsigned int, f);
    u += 0x7fffu + ((u >> 16) & 1u);          // round-to-nearest-even
    return (u16)(u >> 16);
}

// ---------------------------------------------------------------- convert (row-major copy)
__global__ void f32_to_bf16_kernel(const float* __restrict__ in,
                                   u16* __restrict__ out, int n) {
    int i = blockIdx.x * blockDim.x + threadIdx.x;
    int stride = gridDim.x * blockDim.x;
    for (; i < n; i += stride) out[i] = f2bf(in[i]);
}

// ---------------------------------------------------------------- convert + transpose
// in: (K,N) fp32 row-major  ->  out: (N,K) bf16 row-major; batched by z (stride K*N)
__global__ __launch_bounds__(256) void transpose_f32_to_bf16(
    const float* __restrict__ in, u16* __restrict__ out, int K, int N)
{
    __shared__ u16 t[32][33];
    const size_t bz = blockIdx.z;
    in  += bz * (size_t)K * N;
    out += bz * (size_t)K * N;
    int n0 = blockIdx.x * 32, k0 = blockIdx.y * 32;
    int tx = threadIdx.x & 31, ty = threadIdx.x >> 5;     // 32 x 8
#pragma unroll
    for (int j = 0; j < 32; j += 8)
        t[ty + j][tx] = f2bf(in[(size_t)(k0 + ty + j) * N + n0 + tx]);
    __syncthreads();
#pragma unroll
    for (int j = 0; j < 32; j += 8)
        out[(size_t)(n0 + ty + j) * K + k0 + tx] = t[tx][ty + j];
}

// ---------------------------------------------------------------- WMMA GEMM (TDM-staged)
// C[M,N] = A[M,K] * B^T  where B stored as (N,K) bf16 row-major.  Batched by z.
// flags: 2 = relu, 4 = bf16 output, 8 = store C transposed (C[n*ldc + m])
#define BM 64
#define BN 64
#define BK 32
#define LDSS 40           // LDS row stride in u16 (TDM pad: 16 dwords + 4 dwords)
#define TILE_U16 (64 * LDSS)

// loop-invariant descriptor group1: 16x32 bf16 tile, padded to 80B LDS row stride
__device__ __forceinline__ i32x8 tdm_g1_slice(int ld)
{
    i32x8 g1;
    g1[0] = (1 << 16)        // data_size = 2 bytes
          | (1 << 20)        // pad_enable
          | (3 << 22)        // pad_interval: every 16 DWORDs (one 64B row)
          | (3 << 25);       // pad_amount: 4 DWORDs -> LDS row stride 80B
    const unsigned TD = 1u << 20;                 // generous tensor dims (no OOB)
    g1[1] = (int)((TD & 0xFFFFu) << 16);          // tensor_dim0 lo
    g1[2] = (int)((TD >> 16) | ((TD & 0xFFFFu) << 16));   // dim0 hi | dim1 lo
    g1[3] = (int)((TD >> 16) | (32u << 16));      // dim1 hi | tile_dim0 = 32
    g1[4] = 16;                                   // tile_dim1 = 16 (one wave's slice)
    g1[5] = ld;                                   // tensor_dim0_stride[31:0]
    g1[6] = 0;
    g1[7] = 0;
    return g1;
}

__device__ __forceinline__ void tdm_issue(unsigned long long gaddr, unsigned ldsByteOff,
                                          i32x8 g1)
{
    u32x4 g0;
    g0[0] = 1u;                                      // count = 1 (valid)
    g0[1] = ldsByteOff;                              // lds_addr
    g0[2] = (unsigned)gaddr;                         // global_addr[31:0]
    g0[3] = (unsigned)(gaddr >> 32) | (2u << 30);    // global_addr[56:32] | type=2
    i32x4 gz4;
    gz4[0] = 0; gz4[1] = 0; gz4[2] = 0; gz4[3] = 0;
    i32x8 gz8;
#pragma unroll
    for (int i = 0; i < 8; ++i) gz8[i] = 0;
    __builtin_amdgcn_tensor_load_to_lds(g0, g1, gz4, gz4, gz8, 0);
}

__global__ __launch_bounds__(128) void gemm_bf16_wmma(
    const u16* __restrict__ A, const u16* __restrict__ B, void* __restrict__ Cv,
    int M, int N, int K, int lda, int ldb, int ldc,
    const float* __restrict__ bias,
    long long sA, long long sB, long long sC, long long sBias, int flags)
{
    __shared__ u16 smem[4 * TILE_U16];   // [buf0:A, buf0:B, buf1:A, buf1:B]

    const int tid  = threadIdx.x;
    const int lane = tid & 31;
    const int wave = tid >> 5;
    const int half = lane >> 4;          // 0: lanes 0-15, 1: lanes 16-31
    const int mrow = lane & 15;
    const int bz   = blockIdx.z;

    A += (size_t)bz * sA;
    B += (size_t)bz * sB;
    const float* biasP = bias ? (bias + (size_t)bz * sBias) : nullptr;

    const int m0 = blockIdx.y * BM;
    const int n0 = blockIdx.x * BN;
    const bool relu   = (flags & 2) != 0;
    const bool outbf  = (flags & 4) != 0;
    const bool transC = (flags & 8) != 0;

    const unsigned ldsBase = (unsigned)(uintptr_t)(void*)&smem[0];
    const int nk = K / BK;

    // each wave DMA-loads its own 16-row slice of the A and B tiles
    const unsigned sliceOff = (unsigned)(wave * 16 * LDSS * 2);
    const unsigned aLds[2] = { ldsBase + 0u * TILE_U16 * 2 + sliceOff,
                               ldsBase + 2u * TILE_U16 * 2 + sliceOff };
    const unsigned bLds[2] = { ldsBase + 1u * TILE_U16 * 2 + sliceOff,
                               ldsBase + 3u * TILE_U16 * 2 + sliceOff };
    const i32x8 g1a = tdm_g1_slice(lda);
    const i32x8 g1b = tdm_g1_slice(ldb);
    unsigned long long aAddr =
        (unsigned long long)(uintptr_t)(A + (size_t)(m0 + wave * 16) * lda);
    unsigned long long bAddr =
        (unsigned long long)(uintptr_t)(B + (size_t)(n0 + wave * 16) * ldb);

    f32x8 acc[4];
#pragma unroll
    for (int t = 0; t < 4; ++t)
#pragma unroll
        for (int e = 0; e < 8; ++e) acc[t][e] = 0.0f;

    // ---- prologue: every wave DMAs its slice of tile 0 into buffer 0
    tdm_issue(aAddr, aLds[0], g1a);
    tdm_issue(bAddr, bLds[0], g1b);
    __builtin_amdgcn_s_wait_tensorcnt(0);
    __syncthreads();

    unsigned long long aN = aAddr + BK * 2;   // next tile's byte address
    unsigned long long bN = bAddr + BK * 2;

    for (int kt = 0; kt < nk; ++kt) {
        const int cur = kt & 1;
        const int nxt = cur ^ 1;
        // ---- issue DMA for next tile while computing the current one
        if (kt + 1 < nk) {
            tdm_issue(aN, aLds[nxt], g1a);
            tdm_issue(bN, bLds[nxt], g1b);
            aN += BK * 2;
            bN += BK * 2;
        }

        const u16* Asc = smem + (cur * 2 + 0) * TILE_U16;
        const u16* Bsc = smem + (cur * 2 + 1) * TILE_U16;

        // ---- A fragment: lanes 0-15 hold K{0..7,16..23}, lanes 16-31 K{8..15,24..31}
        const int arow = wave * 16 + mrow;
        u16x8 alo = *(const u16x8*)(&Asc[arow * LDSS + half * 8]);
        u16x8 ahi = *(const u16x8*)(&Asc[arow * LDSS + 16 + half * 8]);
        u16x16 afu;
#pragma unroll
        for (int e = 0; e < 8; ++e) { afu[e] = alo[e]; afu[e + 8] = ahi[e]; }
        bf16x16 aF = __builtin_bit_cast(bf16x16, afu);

        // ---- hoist all four B fragments so ds loads clause together
        bf16x16 bF[4];
#pragma unroll
        for (int t = 0; t < 4; ++t) {
            const int brow = t * 16 + mrow;
            u16x8 blo = *(const u16x8*)(&Bsc[brow * LDSS + half * 8]);
            u16x8 bhi = *(const u16x8*)(&Bsc[brow * LDSS + 16 + half * 8]);
            u16x16 bfu;
#pragma unroll
            for (int e = 0; e < 8; ++e) { bfu[e] = blo[e]; bfu[e + 8] = bhi[e]; }
            bF[t] = __builtin_bit_cast(bf16x16, bfu);
        }
#pragma unroll
        for (int t = 0; t < 4; ++t)
            acc[t] = __builtin_amdgcn_wmma_f32_16x16x32_bf16(
                false, aF, false, bF[t], (short)0, acc[t], false, false);

        if (kt + 1 < nk) __builtin_amdgcn_s_wait_tensorcnt(0);
        __syncthreads();
    }

    // ---- epilogue: bias hoisted per column; VGPR i -> M = i + 8*half, N = lane&15
    float bv[4];
#pragma unroll
    for (int t = 0; t < 4; ++t)
        bv[t] = biasP ? biasP[n0 + t * 16 + mrow] : 0.0f;

    float* Cf = (float*)Cv;
    u16*   Cb = (u16*)Cv;
#pragma unroll
    for (int t = 0; t < 4; ++t) {
#pragma unroll
        for (int i = 0; i < 8; ++i) {
            int row = m0 + wave * 16 + i + (half ? 8 : 0);
            int col = n0 + t * 16 + mrow;
            float v = acc[t][i] + bv[t];
            if (relu) v = v > 0.0f ? v : 0.0f;
            size_t off = transC ? ((size_t)bz * sC + (size_t)col * ldc + row)
                                : ((size_t)bz * sC + (size_t)row * ldc + col);
            if (outbf) Cb[off] = f2bf(v); else Cf[off] = v;
        }
    }
    (void)M; (void)N;
}

// ---------------------------------------------------------------- hid = relu(sf0 @ b1w + b1b)
__global__ __launch_bounds__(256) void hid_kernel(
    const float* __restrict__ sf, const float* __restrict__ b1w,
    const float* __restrict__ b1b, float* __restrict__ hid)
{
    int gid = blockIdx.x * 256 + threadIdx.x;     // H*DM = 2048 threads
    int h = gid / DM_, m = gid % DM_;
    float s = b1b[gid];
    for (int k = 0; k < DS_; ++k)
        s += sf[k] * b1w[(size_t)(h * DS_ + k) * DM_ + m];
    hid[gid] = s > 0.0f ? s : 0.0f;
}

// ---------------------------------------------------------------- biasv = hid @ b2w + b2b (streams 512MB)
__global__ __launch_bounds__(256) void attn_bias_kernel(
    const float* __restrict__ hid, const float* __restrict__ b2w,
    const float* __restrict__ b2b, float* __restrict__ biasv)
{
    int h = blockIdx.y;
    int n = blockIdx.x * 256 + threadIdx.x;       // 0..65535
    __shared__ float hs[DM_];
    hs[threadIdx.x] = hid[h * DM_ + threadIdx.x];
    __syncthreads();
    float s = b2b[(size_t)h * (L_ * L_) + n];
#pragma unroll 4
    for (int m = 0; m < DM_; ++m)
        s += hs[m] * b2w[((size_t)(h * DM_ + m)) * (L_ * L_) + n];
    biasv[(size_t)h * (L_ * L_) + n] = s;
}

// ---------------------------------------------------------------- softmax(mask(scores)) + bias -> bf16
__global__ __launch_bounds__(256) void softmax_bias_kernel(
    const float* __restrict__ scores, const int* __restrict__ mask,
    const float* __restrict__ biasv, u16* __restrict__ soft)
{
    int hl = blockIdx.x;            // h*L + l
    int l  = hl & (L_ - 1);
    int n  = threadIdx.x;
    size_t ro = (size_t)hl * L_;
    float s = scores[ro + n] * 0.08838834764831845f;   // 1/sqrt(128)
    if (mask[l * L_ + n] != 0) s = 1e-9f;              // batch-0 mask, pre-softmax
    __shared__ float red[256];
    red[n] = s; __syncthreads();
    for (int o = 128; o > 0; o >>= 1) {
        if (n < o) red[n] = fmaxf(red[n], red[n + o]);
        __syncthreads();
    }
    float mx = red[0]; __syncthreads();
    float e = __expf(s - mx);
    red[n] = e; __syncthreads();
    for (int o = 128; o > 0; o >>= 1) {
        if (n < o) red[n] += red[n + o];
        __syncthreads();
    }
    float p = e / red[0] + biasv[ro + n];
    soft[ro + n] = f2bf(p);
}

// ---------------------------------------------------------------- residual + LayerNorm
__global__ __launch_bounds__(256) void add_ln_kernel(
    const float* __restrict__ base, const float* __restrict__ resid, int residMod,
    const float* __restrict__ g, const float* __restrict__ beta,
    float* __restrict__ outf, u16* __restrict__ outbf)
{
    int bl  = blockIdx.x;
    int tid = threadIdx.x;
    size_t bo = (size_t)bl * D_;
    size_t rb = (size_t)(bl % residMod) * D_;
    float v[4], s = 0.0f;
#pragma unroll
    for (int j = 0; j < 4; ++j) {
        int d = j * 256 + tid;
        v[j] = base[bo + d] + resid[rb + d];
        s += v[j];
    }
    __shared__ float red[256];
    red[tid] = s; __syncthreads();
    for (int o = 128; o > 0; o >>= 1) {
        if (tid < o) red[tid] += red[tid + o];
        __syncthreads();
    }
    float mu = red[0] * (1.0f / D_); __syncthreads();
    s = 0.0f;
#pragma unroll
    for (int j = 0; j < 4; ++j) { float t = v[j] - mu; s += t * t; }
    red[tid] = s; __syncthreads();
    for (int o = 128; o > 0; o >>= 1) {
        if (tid < o) red[tid] += red[tid + o];
        __syncthreads();
    }
    float rstd = rsqrtf(red[0] * (1.0f / D_) + 1e-5f);
#pragma unroll
    for (int j = 0; j < 4; ++j) {
        int d = j * 256 + tid;
        float y = (v[j] - mu) * rstd * g[d] + beta[d];
        outf[bo + d] = y;
        if (outbf) outbf[bo + d] = f2bf(y);
    }
}

// ================================================================ host
extern "C" void kernel_launch(void* const* d_in, const int* in_sizes, int n_in,
                              void* d_out, int out_size, void* d_ws, size_t ws_size,
                              hipStream_t stream) {
    const float* src   = (const float*)d_in[0];
    const float* sf    = (const float*)d_in[1];
    const int*   amask = (const int*)  d_in[2];
    const float* qw = (const float*)d_in[3];  const float* qb = (const float*)d_in[4];
    const float* kw = (const float*)d_in[5];  const float* kb = (const float*)d_in[6];
    const float* vw = (const float*)d_in[7];  const float* vb = (const float*)d_in[8];
    const float* b1w = (const float*)d_in[9];  const float* b1b = (const float*)d_in[10];
    const float* b2w = (const float*)d_in[11]; const float* b2b = (const float*)d_in[12];
    const float* ow = (const float*)d_in[13];  const float* ob = (const float*)d_in[14];
    const float* ln1g = (const float*)d_in[15]; const float* ln1b = (const float*)d_in[16];
    const float* ln2g = (const float*)d_in[17]; const float* ln2b = (const float*)d_in[18];
    const float* w1 = (const float*)d_in[19];  const float* bf1 = (const float*)d_in[20];
    const float* w2 = (const float*)d_in[21];  const float* bf2 = (const float*)d_in[22];
    float* out = (float*)d_out;

    // ---- workspace carve-up
    char* ws = (char*)d_ws;
    size_t off = 0;
    auto alloc = [&](size_t bytes) {
        char* p = ws + off;
        off = (off + bytes + 255) & ~(size_t)255;
        return p;
    };
    u16*   src0b  = (u16*)  alloc((size_t)L_ * D_ * 2);
    u16*   qwT    = (u16*)  alloc((size_t)H_ * D_ * DQ_ * 2);   // (H, DQ, D)
    u16*   kwT    = (u16*)  alloc((size_t)H_ * D_ * DQ_ * 2);
    u16*   vwT    = (u16*)  alloc((size_t)H_ * D_ * DQ_ * 2);
    u16*   qb16   = (u16*)  alloc((size_t)H_ * L_ * DQ_ * 2);   // (H, L, DQ)
    u16*   kb16   = (u16*)  alloc((size_t)H_ * L_ * DQ_ * 2);   // (H, L, DQ)
    u16*   vT     = (u16*)  alloc((size_t)H_ * DQ_ * L_ * 2);   // (H, DQ, L)
    float* scores = (float*)alloc((size_t)H_ * L_ * L_ * 4);
    float* hid    = (float*)alloc((size_t)H_ * DM_ * 4);
    float* biasv  = (float*)alloc((size_t)H_ * L_ * L_ * 4);
    u16*   softb  = (u16*)  alloc((size_t)H_ * L_ * L_ * 2);
    u16*   hidden = (u16*)  alloc((size_t)L_ * H_ * DQ_ * 2);   // (L, H*DQ)
    u16*   owT    = (u16*)  alloc((size_t)D_ * D_ * 2);         // (D_out, H*DQ)
    float* attn   = (float*)alloc((size_t)L_ * D_ * 4);
    float* x      = (float*)alloc((size_t)B_ * L_ * D_ * 4);
    u16*   xb     = (u16*)  alloc((size_t)B_ * L_ * D_ * 2);
    u16*   w1T    = (u16*)  alloc((size_t)D_ * DF_ * 2);        // (DF, D)
    u16*   w2T    = (u16*)  alloc((size_t)DF_ * D_ * 2);        // (D, DF)
    u16*   hmid   = (u16*)  alloc((size_t)B_ * L_ * DF_ * 2);
    float* ffn    = (float*)alloc((size_t)B_ * L_ * D_ * 4);
    (void)ws_size; (void)n_in; (void)in_sizes; (void)out_size;

    dim3 cb(256);
    {   // batch-0 activations, row-major
        int n = L_ * D_;
        f32_to_bf16_kernel<<<dim3((n + 255) / 256), cb, 0, stream>>>(src, src0b, n);
    }
    // weights: convert + transpose so every GEMM consumes B as (N,K)
    transpose_f32_to_bf16<<<dim3(DQ_/32, D_/32, H_), cb, 0, stream>>>(qw, qwT, D_, DQ_);
    transpose_f32_to_bf16<<<dim3(DQ_/32, D_/32, H_), cb, 0, stream>>>(kw, kwT, D_, DQ_);
    transpose_f32_to_bf16<<<dim3(DQ_/32, D_/32, H_), cb, 0, stream>>>(vw, vwT, D_, DQ_);
    transpose_f32_to_bf16<<<dim3(D_/32,  D_/32, 1 ), cb, 0, stream>>>(ow, owT, D_, D_);
    transpose_f32_to_bf16<<<dim3(DF_/32, D_/32, 1 ), cb, 0, stream>>>(w1, w1T, D_, DF_);
    transpose_f32_to_bf16<<<dim3(D_/32,  DF_/32, 1), cb, 0, stream>>>(w2, w2T, DF_, D_);

    dim3 gblk(128);
    // q/k (batch 0): [256x1024] x [1024x128] per head, bf16 out + bias
    gemm_bf16_wmma<<<dim3(DQ_/BN, L_/BM, H_), gblk, 0, stream>>>(
        src0b, qwT, qb16, L_, DQ_, D_, D_, D_, DQ_, qb,
        0, (long long)D_*DQ_, (long long)L_*DQ_, DQ_, /*bf16*/4);
    gemm_bf16_wmma<<<dim3(DQ_/BN, L_/BM, H_), gblk, 0, stream>>>(
        src0b, kwT, kb16, L_, DQ_, D_, D_, D_, DQ_, kb,
        0, (long long)D_*DQ_, (long long)L_*DQ_, DQ_, 4);
    // v: store transposed -> vT (H, DQ, L)
    gemm_bf16_wmma<<<dim3(DQ_/BN, L_/BM, H_), gblk, 0, stream>>>(
        src0b, vwT, vT, L_, DQ_, D_, D_, D_, L_, vb,
        0, (long long)D_*DQ_, (long long)DQ_*L_, DQ_, /*bf16|transC*/12);

    // scores = q @ k^T : kb16 already (N=L, K=DQ)
    gemm_bf16_wmma<<<dim3(L_/BN, L_/BM, H_), gblk, 0, stream>>>(
        qb16, kb16, scores, L_, L_, DQ_, DQ_, DQ_, L_, nullptr,
        (long long)L_*DQ_, (long long)L_*DQ_, (long long)L_*L_, 0, 0);

    hid_kernel<<<dim3((H_*DM_)/256), cb, 0, stream>>>(sf, b1b ? b1w : b1w, b1b, hid);
    attn_bias_kernel<<<dim3((L_*L_)/256, H_), cb, 0, stream>>>(hid, b2w, b2b, biasv);
    softmax_bias_kernel<<<dim3(H_*L_), cb, 0, stream>>>(scores, amask, biasv, softb);

    // head0 = soft @ v : B = vT (N=DQ, K=L); written directly as hidden (L, H*DQ)
    gemm_bf16_wmma<<<dim3(DQ_/BN, L_/BM, H_), gblk, 0, stream>>>(
        softb, vT, hidden, L_, DQ_, L_, L_, L_, H_*DQ_, nullptr,
        (long long)L_*L_, (long long)DQ_*L_, DQ_, 0, 4);

    // attn = hidden @ ow + ob : B = owT (N=D, K=H*DQ)
    gemm_bf16_wmma<<<dim3(D_/BN, L_/BM, 1), gblk, 0, stream>>>(
        hidden, owT, attn, L_, D_, H_*DQ_, H_*DQ_, H_*DQ_, D_, ob,
        0, 0, 0, 0, 0);

    // x = LN1(src + attn), plus bf16 copy
    add_ln_kernel<<<dim3(B_*L_), cb, 0, stream>>>(src, attn, L_, ln1g, ln1b, x, xb);

    // hmid = relu(x @ w1 + bf1) : B = w1T (N=DF, K=D)
    gemm_bf16_wmma<<<dim3(DF_/BN, (B_*L_)/BM, 1), gblk, 0, stream>>>(
        xb, w1T, hmid, B_*L_, DF_, D_, D_, D_, DF_, bf1,
        0, 0, 0, 0, /*relu|bf16*/6);

    // ffn = hmid @ w2 + bf2 : B = w2T (N=D, K=DF)
    gemm_bf16_wmma<<<dim3(D_/BN, (B_*L_)/BM, 1), gblk, 0, stream>>>(
        hmid, w2T, ffn, B_*L_, D_, DF_, DF_, DF_, D_, bf2,
        0, 0, 0, 0, 0);

    // out = LN2(x + ffn)
    add_ln_kernel<<<dim3(B_*L_), cb, 0, stream>>>(x, ffn, B_*L_, ln2g, ln2b, out, nullptr);
}
